// VQVAE_27298812133614
// MI455X (gfx1250) — matrix operand order repaired and checked by reading, not compiled, not run on hardware
//
#include <hip/hip_runtime.h>
#include <hip/hip_bf16.h>
#include <math.h>

// ---------------------------------------------------------------------------
// Types for WMMA fragments (wave32, gfx1250)
// ---------------------------------------------------------------------------
typedef _Float16 v8h  __attribute__((ext_vector_type(8)));
typedef _Float16 v16h __attribute__((ext_vector_type(16)));
typedef float    v8f  __attribute__((ext_vector_type(8)));
typedef float    v4f  __attribute__((ext_vector_type(4)));

#define DEV __device__ __forceinline__

DEV v16h mk16(v8h lo, v8h hi) {
  v16h r;
#pragma unroll
  for (int i = 0; i < 8; ++i) { r[i] = lo[i]; r[i + 8] = hi[i]; }
  return r;
}
DEV v16h zero16() {
  v16h r;
#pragma unroll
  for (int i = 0; i < 16; ++i) r[i] = (_Float16)0.f;
  return r;
}
DEV v8f zero8f() {
  v8f r;
#pragma unroll
  for (int i = 0; i < 8; ++i) r[i] = 0.f;
  return r;
}
// B fragment: lane n holds column co=n0+(lane&15), K chunks [boff..boff+7]
// and [boff+16..boff+23] where boff = (lane<16 ? 0 : 8).  p already includes boff.
DEV v16h load_b16(const _Float16* p) {
  return mk16(*(const v8h*)p, *(const v8h*)(p + 16));
}
// A fragment: lane holds row m=lane&15, K chunks [aoff..aoff+7] and
// [aoff+16..aoff+23] where aoff = (lane<16 ? 0 : 8).  p = row base (+c0).
DEV v16h load_a16(const _Float16* p, int aoff, bool valid) {
  if (!valid) return zero16();
  return mk16(*(const v8h*)(p + aoff), *(const v8h*)(p + aoff + 16));
}
DEV v8f wmma_f16(v16h a, v16h b, v8f c) {
  return __builtin_amdgcn_wmma_f32_16x16x32_f16(false, a, false, b,
                                                (short)0, c, false, false);
}

// ---------------------------------------------------------------------------
// Workspace layout (bytes, 256-aligned)
// ---------------------------------------------------------------------------
static constexpr size_t OFF_ACT_A = 0;          // 33,554,432  f16 acts ping
static constexpr size_t OFF_ACT_B = 33554432;   // 33,554,432  f16 acts pong
static constexpr size_t OFF_RAW   = 67108864;   // 67,108,864  f32 raw conv out
static constexpr size_t OFF_ZF32  = 134217728;  //  4,194,304  z (f32)
static constexpr size_t OFF_E16   = 138412032;  //  1,048,576  codebook f16
static constexpr size_t OFF_E2    = 139460608;  //     16,384  |E|^2
static constexpr size_t OFF_WPE   = 139476992;  //  3,145,728  packed enc W
static constexpr size_t OFF_WPD   = 142622720;  //  3,145,728  packed dec W
static constexpr size_t OFF_SUMS  = 145768448;  //      2,048
static constexpr size_t OFF_SUMSQ = 145770496;  //      2,048
static constexpr size_t OFF_SCALE = 145772544;  //      2,048
static constexpr size_t OFF_SHIFT = 145774592;  //      2,048
static constexpr size_t OFF_IDX   = 145776640;  //     32,768
static constexpr size_t OFF_CNT   = 145809408;  //     16,384  counts
static constexpr size_t OFF_SLOTR = 145825792;  //      1,024  recon partials
static constexpr size_t OFF_SLOTV = 145826816;  //      1,024  vq partials

// ---------------------------------------------------------------------------
// Utility kernels
// ---------------------------------------------------------------------------
__global__ void zero_f32(float* __restrict__ p, int n) {
  int i = blockIdx.x * 256 + threadIdx.x;
  if (i < n) p[i] = 0.f;
}

// x [16][256][4096] f32 (NCL)  ->  x16 [16][4096][256] f16 (N,L,C)
__global__ void transpose_in(const float* __restrict__ x,
                             _Float16* __restrict__ x16) {
  __shared__ float tile[32][33];
  const int C = 256, L = 4096;
  int n = blockIdx.z, l0 = blockIdx.x << 5, c0 = blockIdx.y << 5;
  int tx = threadIdx.x, ty = threadIdx.y;
#pragma unroll
  for (int i = 0; i < 4; ++i) {
    int c = c0 + ty + i * 8;
    tile[ty + i * 8][tx] = x[((size_t)n * C + c) * L + l0 + tx];
  }
  __syncthreads();
#pragma unroll
  for (int i = 0; i < 4; ++i) {
    int l = l0 + ty + i * 8;
    x16[((size_t)n * L + l) * C + c0 + tx] = (_Float16)tile[tx][ty + i * 8];
  }
}

// Pack weights to f16 [k][co][ci].
// tl=0: src = w[co][ci][3] (torch Conv1d);  tl=1: src = w[ci][co][3] (ConvT1d)
// flip=1: source k index = 2-k (used for the stride-1 transposed conv)
__global__ void pack_w(const float* __restrict__ w, _Float16* __restrict__ wp,
                       int CO, int CI, int flip, int tl) {
  int i = blockIdx.x * 256 + threadIdx.x;
  int total = 3 * CO * CI;
  if (i >= total) return;
  int k = i / (CO * CI);
  int r = i - k * CO * CI;
  int co = r / CI;
  int ci = r - co * CI;
  int ks = flip ? (2 - k) : k;
  size_t src = tl ? (((size_t)ci * CO + co) * 3 + ks)
                  : (((size_t)co * CI + ci) * 3 + ks);
  wp[i] = (_Float16)w[src];
}

__global__ void cvt_cb(const float* __restrict__ E, _Float16* __restrict__ e16) {
  int i = blockIdx.x * 256 + threadIdx.x;  // 4096*128 total
  e16[i] = (_Float16)E[i];
}

__global__ void e2_cb(const float* __restrict__ E, float* __restrict__ e2) {
  int c = blockIdx.x * 256 + threadIdx.x;
  if (c < 4096) {
    const float* p = E + (size_t)c * 128;
    float s = 0.f;
    for (int d = 0; d < 128; ++d) { float v = p[d]; s += v * v; }
    e2[c] = s;
  }
}

// ---------------------------------------------------------------------------
// Conv1d (k=3, pad=1, stride s) as implicit GEMM, one wave per 16x16 tile.
// act  : f16 [N][Lin][CI]    wp : f16 [3][CO][CI]
// mode 0: raw f32 [N*Lout][CO] (+bias)
// mode 1: sigmoid -> x_recon f32 NCL [N][CO][Lout] + sq-err partials vs xin
// ---------------------------------------------------------------------------
__global__ __launch_bounds__(32) void conv3_wmma(
    const _Float16* __restrict__ act, const _Float16* __restrict__ wp,
    const float* __restrict__ bias, float* __restrict__ raw,
    const float* __restrict__ xin, float* __restrict__ xrec,
    float* __restrict__ slot,
    int N, int CI, int CO, int Lin, int Lout, int stride, int mode) {
  int lane = (int)threadIdx.x;
  int ltiles = Lout >> 4;
  int n  = blockIdx.x / ltiles;
  int l0 = (blockIdx.x % ltiles) << 4;
  int co0 = blockIdx.y << 4;
  int m = lane & 15;
  int hi = lane >> 4;
  int aoff = hi ? 8 : 0;
  int boff = hi ? 8 : 0;
  v8f acc = zero8f();

  for (int k = 0; k < 3; ++k) {
    int li = (l0 + m) * stride + k - 1;
    bool valid = (li >= 0) && (li < Lin);
    int lic = valid ? li : 0;
    const _Float16* ap = act + ((size_t)n * Lin + lic) * CI;
    const _Float16* bp = wp + ((size_t)k * CO + co0 + m) * CI + boff;
    for (int c0 = 0; c0 < CI; c0 += 32) {
      v16h a = load_a16(ap + c0, aoff, valid);
      v16h b = load_b16(bp + c0);
      acc = wmma_f16(a, b, acc);
    }
  }
  float bv = bias[co0 + m];

  if (mode == 0) {
#pragma unroll
    for (int v = 0; v < 8; ++v) {
      int row = l0 + v + (hi ? 8 : 0);
      raw[((size_t)n * Lout + row) * CO + co0 + m] = acc[v] + bv;
    }
  } else {
    // D frag: fixed co per lane, 8 consecutive l values -> contiguous in NCL
    int c = co0 + m;
    int lbase = l0 + (hi ? 8 : 0);
    const float* xp = xin + ((size_t)n * CO + c) * (size_t)Lout + lbase;
    float* op = xrec + ((size_t)n * CO + c) * (size_t)Lout + lbase;
    float vals[8];
    float lsum = 0.f;
#pragma unroll
    for (int v = 0; v < 8; ++v) {
      float s = 1.f / (1.f + expf(-(acc[v] + bv)));
      vals[v] = s;
      float d = s - xp[v];
      lsum += d * d;
    }
    v4f s0, s1;
#pragma unroll
    for (int i = 0; i < 4; ++i) { s0[i] = vals[i]; s1[i] = vals[i + 4]; }
    *(v4f*)op = s0;
    *(v4f*)(op + 4) = s1;
#pragma unroll
    for (int off = 16; off >= 1; off >>= 1) lsum += __shfl_xor(lsum, off, 32);
    if (lane == 0) atomicAdd(&slot[blockIdx.x & 255], lsum);
  }
}

// ---------------------------------------------------------------------------
// ConvTranspose1d (k=3, stride 2, pad 1, out_pad 1):
//   y[2s]   = W[:,:,1]^T x[s]
//   y[2s+1] = W[:,:,2]^T x[s] + W[:,:,0]^T x[s+1]
// wp : f16 [k][co][ci] with wp[k][co][ci] = w[ci][co][k]
// ---------------------------------------------------------------------------
__global__ __launch_bounds__(32) void convt3_wmma(
    const _Float16* __restrict__ act, const _Float16* __restrict__ wp,
    const float* __restrict__ bias, float* __restrict__ raw,
    int N, int CI, int CO, int Ls) {
  int lane = (int)threadIdx.x;
  int stiles = Ls >> 4;
  int n  = blockIdx.x / stiles;
  int s0 = (blockIdx.x % stiles) << 4;
  int co0 = blockIdx.y << 4;
  int m = lane & 15;
  int hi = lane >> 4;
  int aoff = hi ? 8 : 0;
  int boff = hi ? 8 : 0;
  v8f accE = zero8f(), accO = zero8f();

  int s = s0 + m;
  bool v1 = (s + 1) < Ls;
  int s1 = v1 ? (s + 1) : s;
  const _Float16* a0p = act + ((size_t)n * Ls + s) * CI;
  const _Float16* a1p = act + ((size_t)n * Ls + s1) * CI;
  const _Float16* b0p = wp + ((size_t)0 * CO + co0 + m) * CI + boff;
  const _Float16* b1p = wp + ((size_t)1 * CO + co0 + m) * CI + boff;
  const _Float16* b2p = wp + ((size_t)2 * CO + co0 + m) * CI + boff;

  for (int c0 = 0; c0 < CI; c0 += 32) {
    v16h a0 = load_a16(a0p + c0, aoff, true);
    v16h a1 = load_a16(a1p + c0, aoff, v1);
    v16h b1 = load_b16(b1p + c0);
    v16h b2 = load_b16(b2p + c0);
    v16h b0 = load_b16(b0p + c0);
    accE = wmma_f16(a0, b1, accE);
    accO = wmma_f16(a0, b2, accO);
    accO = wmma_f16(a1, b0, accO);
  }
  float bv = bias[co0 + m];
  int Lout = Ls << 1;
#pragma unroll
  for (int v = 0; v < 8; ++v) {
    int srow = s0 + v + (hi ? 8 : 0);
    size_t base = ((size_t)n * Lout + 2 * srow) * CO + co0 + m;
    raw[base] = accE[v] + bv;
    raw[base + CO] = accO[v] + bv;
  }
}

// ---------------------------------------------------------------------------
// BatchNorm helpers (train mode: stats over N*L rows per channel)
// ---------------------------------------------------------------------------
__global__ void bn_reduce(const float* __restrict__ raw, float* __restrict__ sums,
                          float* __restrict__ sumsq, int R, int C) {
  int tid = (int)threadIdx.x;
  int rpb = R >> 7;  // grid.x == 128
  int r0 = blockIdx.x * rpb;
  for (int c = tid; c < C; c += 256) {
    float s = 0.f, q = 0.f;
    const float* p = raw + (size_t)r0 * C + c;
    for (int r = 0; r < rpb; ++r) {
      float v = p[(size_t)r * C];
      s += v; q += v * v;
    }
    atomicAdd(&sums[c], s);
    atomicAdd(&sumsq[c], q);
  }
}

__global__ void bn_final(const float* __restrict__ sums, const float* __restrict__ sumsq,
                         const float* __restrict__ g, const float* __restrict__ be,
                         float* __restrict__ scale, float* __restrict__ shift,
                         int C, float R) {
  int c = blockIdx.x * 256 + (int)threadIdx.x;
  if (c < C) {
    float m = sums[c] / R;
    float v = sumsq[c] / R - m * m;
    float sc = g[c] * rsqrtf(v + 1e-5f);
    scale[c] = sc;
    shift[c] = be[c] - m * sc;
  }
}

__global__ void bn_apply(const float* __restrict__ raw, const float* __restrict__ scale,
                         const float* __restrict__ shift, _Float16* __restrict__ act,
                         float* __restrict__ zout, int total, int cmask) {
  int i = blockIdx.x * 256 + (int)threadIdx.x;
  if (i >= total) return;
  int c = i & cmask;
  float y = raw[i] * scale[c] + shift[c];
  y = (y > 0.f) ? y : 0.01f * y;
  act[i] = (_Float16)y;
  if (zout) zout[i] = y;
}

// ---------------------------------------------------------------------------
// VQ: distances via WMMA (only e2 - 2*dot matters for argmin), running
// per-lane argmin over 256 code tiles, 16-lane shuffle reduce w/ tie-break.
// ---------------------------------------------------------------------------
__global__ __launch_bounds__(32) void vq_argmin(
    const _Float16* __restrict__ z16, const _Float16* __restrict__ e16,
    const float* __restrict__ e2, int* __restrict__ idx,
    float* __restrict__ counts) {
  int lane = (int)threadIdx.x;
  int r0 = blockIdx.x << 4;
  int m = lane & 15;
  int hi = lane >> 4;
  int aoff = hi ? 8 : 0;
  int boff = hi ? 8 : 0;

  const _Float16* zr = z16 + (size_t)(r0 + m) * 128;
  v16h za[4];
#pragma unroll
  for (int cc = 0; cc < 4; ++cc) za[cc] = load_a16(zr + cc * 32, aoff, true);

  float bestV[8];
  int bestI[8];
#pragma unroll
  for (int v = 0; v < 8; ++v) { bestV[v] = 3.4e38f; bestI[v] = 0; }

  for (int j0 = 0; j0 < 4096; j0 += 16) {
    v8f acc = zero8f();
    const _Float16* ep = e16 + (size_t)(j0 + m) * 128 + boff;
#pragma unroll
    for (int cc = 0; cc < 4; ++cc)
      acc = wmma_f16(za[cc], load_b16(ep + cc * 32), acc);
    float e2v = e2[j0 + m];
    int ci = j0 + m;
#pragma unroll
    for (int v = 0; v < 8; ++v) {
      float d = e2v - 2.f * acc[v];
      if (d < bestV[v]) { bestV[v] = d; bestI[v] = ci; }
    }
  }
  // reduce across the 16 lanes of each half-wave (rows are independent halves)
#pragma unroll
  for (int v = 0; v < 8; ++v) {
    float bv = bestV[v];
    int bi = bestI[v];
#pragma unroll
    for (int off = 1; off < 16; off <<= 1) {
      float ov = __shfl_xor(bv, off, 32);
      int oi = __shfl_xor(bi, off, 32);
      if (ov < bv || (ov == bv && oi < bi)) { bv = ov; bi = oi; }
    }
    bestV[v] = bv; bestI[v] = bi;
  }
  if (m == 0) {  // lanes 0 (rows r0..r0+7) and 16 (rows r0+8..r0+15)
    int rbase = r0 + (hi ? 8 : 0);
#pragma unroll
    for (int v = 0; v < 8; ++v) {
      idx[rbase + v] = bestI[v];
      atomicAdd(&counts[bestI[v]], 1.f);
    }
  }
}

// gather codewords: q16 [8192][128] f16 (decoder input), quantized_out f32
// [16][128][512] (NCL transpose), vq squared-error partials
__global__ void vq_gather(const int* __restrict__ idx, const float* __restrict__ E,
                          const float* __restrict__ zf, _Float16* __restrict__ q16,
                          float* __restrict__ qout, float* __restrict__ slotV) {
  int row = blockIdx.x;
  int d = (int)threadIdx.x;  // 128 threads
  int id = idx[row];
  float q = E[(size_t)id * 128 + d];
  float z = zf[(size_t)row * 128 + d];
  q16[(size_t)row * 128 + d] = (_Float16)q;
  int n = row >> 9, t = row & 511;
  qout[((size_t)n * 128 + d) * 512 + t] = q;
  float diff = q - z;
  float s = diff * diff;
#pragma unroll
  for (int off = 16; off >= 1; off >>= 1) s += __shfl_xor(s, off, 32);
  if ((d & 31) == 0) atomicAdd(&slotV[blockIdx.x & 255], s);
}

// ---------------------------------------------------------------------------
// Final scalars: total_loss, recon_loss, vq_loss, perplexity
// ---------------------------------------------------------------------------
__global__ void final_scalars(const float* __restrict__ slotR,
                              const float* __restrict__ slotV,
                              const float* __restrict__ counts,
                              float* __restrict__ outs) {
  __shared__ float red[256];
  int t = (int)threadIdx.x;
  red[t] = slotR[t];
  __syncthreads();
  for (int s = 128; s > 0; s >>= 1) { if (t < s) red[t] += red[t + s]; __syncthreads(); }
  float reconSum = red[0];
  __syncthreads();
  red[t] = slotV[t];
  __syncthreads();
  for (int s = 128; s > 0; s >>= 1) { if (t < s) red[t] += red[t + s]; __syncthreads(); }
  float vqSum = red[0];
  __syncthreads();
  float h = 0.f;
  for (int k = t; k < 4096; k += 256) {
    float p = counts[k] * (1.f / 8192.f);
    h += p * logf(p + 1e-10f);
  }
  red[t] = h;
  __syncthreads();
  for (int s = 128; s > 0; s >>= 1) { if (t < s) red[t] += red[t + s]; __syncthreads(); }
  if (t == 0) {
    float recon = reconSum * (1.f / 16777216.f);   // 16*256*4096
    float vq = 1.25f * vqSum * (1.f / 1048576.f);  // 16*512*128
    outs[0] = recon + vq;
    outs[1] = recon;
    outs[2] = vq;
    outs[3] = expf(-red[0]);
  }
}

// ---------------------------------------------------------------------------
// Host orchestration
// ---------------------------------------------------------------------------
extern "C" void kernel_launch(void* const* d_in, const int* in_sizes, int n_in,
                              void* d_out, int out_size, void* d_ws, size_t ws_size,
                              hipStream_t stream) {
  (void)in_sizes; (void)n_in; (void)out_size; (void)ws_size;
  char* ws = (char*)d_ws;
  const float* x = (const float*)d_in[0];
  const float* cb = (const float*)d_in[17];
  float* out = (float*)d_out;

  _Float16* actA = (_Float16*)(ws + OFF_ACT_A);
  _Float16* actB = (_Float16*)(ws + OFF_ACT_B);
  float* raw    = (float*)(ws + OFF_RAW);
  float* zf32   = (float*)(ws + OFF_ZF32);
  _Float16* e16 = (_Float16*)(ws + OFF_E16);
  float* e2p    = (float*)(ws + OFF_E2);
  _Float16* wpe = (_Float16*)(ws + OFF_WPE);
  _Float16* wpd = (_Float16*)(ws + OFF_WPD);
  float* sums   = (float*)(ws + OFF_SUMS);
  float* sumsq  = (float*)(ws + OFF_SUMSQ);
  float* scale  = (float*)(ws + OFF_SCALE);
  float* shift  = (float*)(ws + OFF_SHIFT);
  int*   idxp   = (int*)(ws + OFF_IDX);
  float* counts = (float*)(ws + OFF_CNT);
  float* slotR  = (float*)(ws + OFF_SLOTR);
  float* slotV  = (float*)(ws + OFF_SLOTV);

  // counts + slotR + slotV are contiguous: 4096+256+256 floats
  zero_f32<<<18, 256, 0, stream>>>(counts, 4608);

  // input NCL f32 -> [N][L][C] f16
  transpose_in<<<dim3(128, 8, 16), dim3(32, 8), 0, stream>>>(x, actA);

  // pack weights
  const size_t ewoff[4] = {0, 196608, 589824, 1376256};
  const int eci[4] = {256, 256, 512, 512};
  const int eco[4] = {256, 512, 512, 128};
  for (int i = 0; i < 4; ++i) {
    int tot = 3 * eco[i] * eci[i];
    pack_w<<<(tot + 255) / 256, 256, 0, stream>>>(
        (const float*)d_in[1 + 4 * i], wpe + ewoff[i], eco[i], eci[i], 0, 0);
  }
  const size_t dwoff[4] = {0, 196608, 983040, 1376256};
  const int dci[4] = {128, 512, 512, 256};
  const int dco[4] = {512, 512, 256, 256};
  const int dwi[4] = {18, 22, 26, 30};
  for (int i = 0; i < 4; ++i) {
    int tot = 3 * dco[i] * dci[i];
    pack_w<<<(tot + 255) / 256, 256, 0, stream>>>(
        (const float*)d_in[dwi[i]], wpd + dwoff[i], dco[i], dci[i],
        (i == 3) ? 1 : 0, 1);
  }
  cvt_cb<<<2048, 256, 0, stream>>>(cb, e16);
  e2_cb<<<16, 256, 0, stream>>>(cb, e2p);

  // ----- encoder -----
  const int elin[4]  = {4096, 2048, 1024, 512};
  const int elout[4] = {2048, 1024, 512, 512};
  const int estr[4]  = {2, 2, 2, 1};
  _Float16* cur = actA;
  _Float16* nxt = actB;
  for (int i = 0; i < 4; ++i) {
    int CI = eci[i], CO = eco[i], Lin = elin[i], Lout = elout[i];
    conv3_wmma<<<dim3((Lout / 16) * 16, CO / 16), 32, 0, stream>>>(
        cur, wpe + ewoff[i], (const float*)d_in[2 + 4 * i], raw,
        nullptr, nullptr, nullptr, 16, CI, CO, Lin, Lout, estr[i], 0);
    int R = 16 * Lout;
    zero_f32<<<4, 256, 0, stream>>>(sums, 1024);
    bn_reduce<<<128, 256, 0, stream>>>(raw, sums, sumsq, R, CO);
    bn_final<<<2, 256, 0, stream>>>(sums, sumsq, (const float*)d_in[3 + 4 * i],
                                    (const float*)d_in[4 + 4 * i], scale, shift,
                                    CO, (float)R);
    int tot = R * CO;
    bn_apply<<<(tot + 255) / 256, 256, 0, stream>>>(
        raw, scale, shift, nxt, (i == 3) ? zf32 : nullptr, tot, CO - 1);
    _Float16* t = cur; cur = nxt; nxt = t;
  }
  // cur == actA holds z16 [8192][128]; zf32 holds z (f32)

  // ----- vector quantizer -----
  vq_argmin<<<512, 32, 0, stream>>>(cur, e16, e2p, idxp, counts);
  vq_gather<<<8192, 128, 0, stream>>>(idxp, cb, zf32, actB, out + 16777216, slotV);

  // ----- decoder (3 stride-2 transposed convs) -----
  cur = actB; nxt = actA;
  const int dls[3] = {512, 1024, 2048};
  const int dgi[3] = {20, 24, 28};
  for (int i = 0; i < 3; ++i) {
    int CI = dci[i], CO = dco[i], Ls = dls[i];
    convt3_wmma<<<dim3((Ls / 16) * 16, CO / 16), 32, 0, stream>>>(
        cur, wpd + dwoff[i], (const float*)d_in[dwi[i] + 1], raw, 16, CI, CO, Ls);
    int R = 16 * Ls * 2;
    zero_f32<<<4, 256, 0, stream>>>(sums, 1024);
    bn_reduce<<<128, 256, 0, stream>>>(raw, sums, sumsq, R, CO);
    bn_final<<<2, 256, 0, stream>>>(sums, sumsq, (const float*)d_in[dgi[i]],
                                    (const float*)d_in[dgi[i] + 1], scale, shift,
                                    CO, (float)R);
    int tot = R * CO;
    bn_apply<<<(tot + 255) / 256, 256, 0, stream>>>(raw, scale, shift, nxt,
                                                    nullptr, tot, CO - 1);
    _Float16* t = cur; cur = nxt; nxt = t;
  }
  // cur == actA: last hidden [16][4096][256] f16

  // ----- final conv (stride-1 transposed == conv with flipped W) + sigmoid -----
  conv3_wmma<<<dim3((4096 / 16) * 16, 16), 32, 0, stream>>>(
      cur, wpd + dwoff[3], (const float*)d_in[31], nullptr,
      x, out, slotR, 16, 256, 256, 4096, 4096, 1, 1);

  final_scalars<<<1, 256, 0, stream>>>(slotR, slotV, counts, out + 17825792);
}
